// GRUModel_15728170238726
// MI455X (gfx1250) — matrix operand order repaired and checked by reading, not compiled, not run on hardware
//
#include <hip/hip_runtime.h>
#include <hip/hip_bf16.h>

// ---------------------------------------------------------------------------
// Bidirectional 2-layer GRU (B=256, T=2048, IN=29, H=64) + LN + MLP head.
// CDNA5 (gfx1250, wave32):
//   - Orientation: D = W_tile(A, 16 gates x K) x Act(B, K x 16 batch).
//     One wave owns gates r,z,n for 16 hidden units => no inter-wave gate
//     exchange; hidden state in registers; ONE barrier per step via
//     double-buffered LDS h broadcast (loop unrolled x2 -> static buffers).
//   - x B-fragments software-pipelined one timestep ahead (ping-pong regs).
//   - 16 batch tiles x 2 directions = 32 WGs, 4 waves each (1 per SIMD).
//   - v_wmma_f32_16x16x32_bf16; weights pinned in VGPRs.
//   - x pre-padded/converted to bf16 [B,T,32] once; out0 stored bf16.
//   - native v_tanh_f32 / raw v_rcp_f32 for gate nonlinearities.
// ---------------------------------------------------------------------------

typedef __bf16 bf16;
typedef __attribute__((ext_vector_type(16))) __bf16 v16bf;
typedef __attribute__((ext_vector_type(8)))  __bf16 v8bf;
typedef __attribute__((ext_vector_type(8)))  float  v8f;

union BF16Frag { v16bf v; v8bf h2[2]; };

static constexpr int Bn   = 256;
static constexpr int Tn   = 2048;
static constexpr int INn  = 29;
static constexpr int Hn   = 64;
static constexpr int OUTn = 11;
static constexpr int XPAD = 32;       // padded input features (bf16)

// ---------------------------------------------------------------------------
__device__ __forceinline__ v8f wmma_bf16(v16bf a, v16bf b, v8f c) {
  return __builtin_amdgcn_wmma_f32_16x16x32_bf16(
      false, a, false, b, (short)0, c, false, false);
}

__device__ __forceinline__ float fast_sigmoid(float x) {
  return __builtin_amdgcn_rcpf(1.0f + __expf(-x));   // raw v_rcp_f32
}

__device__ __forceinline__ float fast_tanh(float x) {
#if __has_builtin(__builtin_amdgcn_tanhf)
  return __builtin_amdgcn_tanhf(x);                  // gfx1250 V_TANH_F32
#else
  const float e = __expf(2.0f * x);
  return 1.0f - 2.0f * __builtin_amdgcn_rcpf(e + 1.0f);
#endif
}

// ---------------------------------------------------------------------------
// A-fragment (16x32 bf16) from row-major f32 weights: A[m][k] = W[g0+m][k].
// ISA A layout: lanes 0-15 (M=lane): halves 0-7 = K klo..klo+7, halves 8-15 =
// K klo+16..klo+23, klo = 0 (8 for lanes>=16).  Built once, pinned in VGPRs.
// ---------------------------------------------------------------------------
__device__ __forceinline__ v16bf make_a_w(const float* __restrict__ W,
                                          int ldW, int g0, int kb,
                                          int kmax, int lane) {
  const int m   = lane & 15;
  const int klo = kb * 32 + ((lane & 16) ? 8 : 0);
  const float* wrow = W + (size_t)(g0 + m) * ldW;
  BF16Frag f;
#pragma unroll
  for (int hh = 0; hh < 16; ++hh) {
    const int k = klo + (hh & 7) + ((hh & 8) ? 16 : 0);
    f.h2[hh >> 3][hh & 7] = (k < kmax) ? (bf16)wrow[k] : (bf16)0.0f;
  }
  return f.v;
}

// ---------------------------------------------------------------------------
// B-fragment (32x16 bf16) from bf16 activations, batch-row-major:
// B[k][n] = act[row n][k].  Lane n (0-15) holds K 0..7,16..23, lanes 16-31
// hold K 8..15,24..31 -> two 16-byte loads per lane (LDS or global).
// ---------------------------------------------------------------------------
__device__ __forceinline__ v16bf load_b_act(const bf16* __restrict__ base,
                                            size_t rowStride, int kb, int lane) {
  const int n  = lane & 15;
  const int ks = kb * 32 + ((lane & 16) ? 8 : 0);
  const bf16* p = base + (size_t)n * rowStride + ks;
  BF16Frag f;
  f.h2[0] = *(const v8bf*)(p);
  f.h2[1] = *(const v8bf*)(p + 16);
  return f.v;
}

// ---------------------------------------------------------------------------
// One-shot pad/convert: x f32 [B,T,29] -> xpad bf16 [B,T,32] (zero padded).
// ---------------------------------------------------------------------------
__global__ __launch_bounds__(256) void pad_x_kernel(const float* __restrict__ x,
                                                    bf16* __restrict__ xp) {
  const size_t i = (size_t)blockIdx.x * 256 + threadIdx.x;   // over B*T*32
  const int    k   = (int)(i & (XPAD - 1));
  const size_t row = i >> 5;
  const float  v   = (k < INn) ? x[row * INn + k] : 0.0f;
  xp[i] = (bf16)v;
}

// ---------------------------------------------------------------------------
// GRU recurrence layer.  4 waves; wave j owns hidden units [16j,16j+16),
// all three gates.  KBX = # 32-wide K blocks of the input.
// ---------------------------------------------------------------------------
template <int KBX, bool STORE_SEQ, bool STORE_FIN>
__global__ __launch_bounds__(128) void gru_rec(
    const bf16* __restrict__ xin,     // bf16 [B][T][KBX*32]
    const float* __restrict__ WihF, const float* __restrict__ WhhF,
    const float* __restrict__ bihF, const float* __restrict__ bhhF,
    const float* __restrict__ WihB, const float* __restrict__ WhhB,
    const float* __restrict__ bihB, const float* __restrict__ bhhB,
    int kmaxX,                        // valid K of input (29 or 128)
    bf16* __restrict__ outSeq,        // bf16 [B][T][2H]  (if STORE_SEQ)
    float* __restrict__ hT)           // f32  [2][B][H]   (if STORE_FIN)
{
  const int lane = threadIdx.x & 31;
  const int wave = threadIdx.x >> 5;        // 0..3 : hidden-unit tile
  const int b0   = blockIdx.x << 4;         // batch tile base
  const int dir  = blockIdx.y;              // 0 fwd, 1 bwd
  const int XW   = KBX * 32;                // padded input width

  const float* Wih = dir ? WihB : WihF;
  const float* Whh = dir ? WhhB : WhhF;
  const float* bih = dir ? bihB : bihF;
  const float* bhh = dir ? bhhB : bhhF;

  // Double-buffered hidden state broadcast (batch-row-major [16][64] bf16).
  __shared__ __align__(16) bf16 hbuf[2][16][Hn];
  for (int i = threadIdx.x; i < 2 * 16 * Hn; i += 128)
    ((bf16*)hbuf)[i] = (bf16)0.0f;

  const int u0 = wave << 4;                 // first hidden unit of this wave
  const int nc = lane & 15;                 // batch column
  const int hi = (lane >> 4) & 1;           // row-half select

  // Weight A-fragments, pinned in VGPRs.
  v16bf wx_r[KBX], wx_z[KBX], wx_n[KBX];
#pragma unroll
  for (int kb = 0; kb < KBX; ++kb) {
    wx_r[kb] = make_a_w(Wih, kmaxX, u0,          kb, kmaxX, lane);
    wx_z[kb] = make_a_w(Wih, kmaxX, Hn + u0,     kb, kmaxX, lane);
    wx_n[kb] = make_a_w(Wih, kmaxX, 2 * Hn + u0, kb, kmaxX, lane);
  }
  v16bf wh_r[2], wh_z[2], wh_n[2];
#pragma unroll
  for (int kb = 0; kb < 2; ++kb) {
    wh_r[kb] = make_a_w(Whh, Hn, u0,          kb, Hn, lane);
    wh_z[kb] = make_a_w(Whh, Hn, Hn + u0,     kb, Hn, lane);
    wh_n[kb] = make_a_w(Whh, Hn, 2 * Hn + u0, kb, Hn, lane);
  }

  // Per-lane bias vectors: rows (gates) v + 8*hi of this wave's tile.
  const int gr = u0 + (hi << 3);
  v8f br, bz, bnx, bnh;
#pragma unroll
  for (int v = 0; v < 8; ++v) {
    br[v]  = bih[gr + v]          + bhh[gr + v];
    bz[v]  = bih[Hn + gr + v]     + bhh[Hn + gr + v];
    bnx[v] = bih[2 * Hn + gr + v];
    bnh[v] = bhh[2 * Hn + gr + v];
  }

  // Hidden state in registers: element (unit = u0+8*hi+v, batch = nc).
  v8f hreg;
#pragma unroll
  for (int v = 0; v < 8; ++v) hreg[v] = 0.0f;

  // Running pointers (t advances by +/-1 each step).
  const int   t0   = dir ? (Tn - 1) : 0;
  const long  xdel = (long)(dir ? -XW : XW);
  const size_t xRowStride = (size_t)Tn * XW;
  const bf16* xptr = xin + ((size_t)b0 * Tn + t0) * XW;      // lane row added in loader
  bf16* outptr = nullptr;
  long  odel   = 0;
  if (STORE_SEQ) {
    outptr = outSeq + ((size_t)(b0 + nc) * Tn + t0) * (2 * Hn)
                    + dir * Hn + u0 + (hi << 3);
    odel   = (long)(dir ? -(2 * Hn) : (2 * Hn));
  }
  const long pfOff = (size_t)nc * xRowStride;   // this lane's row offset, for prefetch

  // One GRU step: read h from hrd, consume x fragments xb, write h to hwr.
  auto phase = [&](const bf16* hrd, bf16* hwr, const v16bf (&xb)[KBX]) {
    const v16bf hb0 = load_b_act(hrd, Hn, 0, lane);
    const v16bf hb1 = load_b_act(hrd, Hn, 1, lane);

    v8f ar = br, az = bz, anx = bnx, anh = bnh;
#pragma unroll
    for (int kb = 0; kb < KBX; ++kb) {
      ar  = wmma_bf16(wx_r[kb], xb[kb], ar);
      az  = wmma_bf16(wx_z[kb], xb[kb], az);
      anx = wmma_bf16(wx_n[kb], xb[kb], anx);
    }
    ar  = wmma_bf16(wh_r[0], hb0, ar);
    az  = wmma_bf16(wh_z[0], hb0, az);
    anh = wmma_bf16(wh_n[0], hb0, anh);
    ar  = wmma_bf16(wh_r[1], hb1, ar);
    az  = wmma_bf16(wh_z[1], hb1, az);
    anh = wmma_bf16(wh_n[1], hb1, anh);

    v8bf hv;
#pragma unroll
    for (int v = 0; v < 8; ++v) {
      const float r  = fast_sigmoid(ar[v]);
      const float z  = fast_sigmoid(az[v]);
      const float ng = fast_tanh(anx[v] + r * anh[v]);
      const float hn = (1.0f - z) * ng + z * hreg[v];
      hreg[v] = hn;
      hv[v]   = (bf16)hn;
    }

    *(v8bf*)(hwr + (size_t)nc * Hn + u0 + (hi << 3)) = hv;   // ds_store_b128
    if (STORE_SEQ) {
      *(v8bf*)outptr = hv;                                   // global_store_b128
      outptr += odel;
    }
  };

  // Preload x fragments for t0.
  v16bf xbA[KBX], xbB[KBX];
#pragma unroll
  for (int kb = 0; kb < KBX; ++kb) xbA[kb] = load_b_act(xptr, xRowStride, kb, lane);

  __syncthreads();

  for (int s = 0; s < Tn; s += 2) {
    // -- step s: h[0] -> h[1]; consume xbA; preload xbB for step s+1 --------
    xptr += xdel;
#pragma unroll
    for (int kb = 0; kb < KBX; ++kb) xbB[kb] = load_b_act(xptr, xRowStride, kb, lane);
    __builtin_prefetch(xptr + xdel + pfOff, 0, 0);           // t+2 row
    phase(&hbuf[0][0][0], &hbuf[1][0][0], xbA);
    __syncthreads();

    // -- step s+1: h[1] -> h[0]; consume xbB; preload xbA for step s+2 ------
    if (s + 2 < Tn) xptr += xdel;     // clamp on last pair (dead reload)
#pragma unroll
    for (int kb = 0; kb < KBX; ++kb) xbA[kb] = load_b_act(xptr, xRowStride, kb, lane);
    phase(&hbuf[1][0][0], &hbuf[0][0][0], xbB);
    __syncthreads();
  }

  if (STORE_FIN) {     // final hidden state straight from registers (f32)
    float* dst = hT + (size_t)dir * Bn * Hn + (size_t)(b0 + nc) * Hn
                    + u0 + (hi << 3);
    *(v8f*)dst = hreg;
  }
}

// ---------------------------------------------------------------------------
// Head: emb = [hTf | hTb] (128) -> LayerNorm -> FC(128->64)+ReLU -> FC(64->11)
// ---------------------------------------------------------------------------
__global__ __launch_bounds__(128) void gru_head(
    const float* __restrict__ hT,
    const float* __restrict__ ln_g, const float* __restrict__ ln_b,
    const float* __restrict__ W1, const float* __restrict__ b1,
    const float* __restrict__ W2, const float* __restrict__ b2,
    float* __restrict__ out)
{
  const int b   = blockIdx.x;
  const int tid = threadIdx.x;                 // 0..127

  __shared__ float y[128];
  __shared__ float red[128];
  __shared__ float h1[64];

  const float ev = (tid < Hn)
      ? hT[(size_t)b * Hn + tid]
      : hT[(size_t)Bn * Hn + (size_t)b * Hn + (tid - Hn)];

  red[tid] = ev;
  __syncthreads();
#pragma unroll
  for (int o = 64; o > 0; o >>= 1) {
    if (tid < o) red[tid] += red[tid + o];
    __syncthreads();
  }
  const float mu = red[0] * (1.0f / 128.0f);
  __syncthreads();

  const float d = ev - mu;
  red[tid] = d * d;
  __syncthreads();
#pragma unroll
  for (int o = 64; o > 0; o >>= 1) {
    if (tid < o) red[tid] += red[tid + o];
    __syncthreads();
  }
  const float var = red[0] * (1.0f / 128.0f);

  y[tid] = d * __frsqrt_rn(var + 1e-5f) * ln_g[tid] + ln_b[tid];
  __syncthreads();

  if (tid < Hn) {
    float acc = b1[tid];
    const float* w = W1 + (size_t)tid * 128;
#pragma unroll 8
    for (int k = 0; k < 128; ++k) acc += y[k] * w[k];
    h1[tid] = fmaxf(acc, 0.0f);
  }
  __syncthreads();

  if (tid < OUTn) {
    float acc = b2[tid];
    const float* w = W2 + (size_t)tid * Hn;
#pragma unroll 8
    for (int k = 0; k < Hn; ++k) acc += h1[k] * w[k];
    out[(size_t)b * OUTn + tid] = acc;
  }
}

// ---------------------------------------------------------------------------
// Launch
// ---------------------------------------------------------------------------
extern "C" void kernel_launch(void* const* d_in, const int* in_sizes, int n_in,
                              void* d_out, int out_size, void* d_ws, size_t ws_size,
                              hipStream_t stream) {
  const float* x      = (const float*)d_in[0];
  const float* Wih00  = (const float*)d_in[1];
  const float* Whh00  = (const float*)d_in[2];
  const float* bih00  = (const float*)d_in[3];
  const float* bhh00  = (const float*)d_in[4];
  const float* Wih01  = (const float*)d_in[5];
  const float* Whh01  = (const float*)d_in[6];
  const float* bih01  = (const float*)d_in[7];
  const float* bhh01  = (const float*)d_in[8];
  const float* Wih10  = (const float*)d_in[9];
  const float* Whh10  = (const float*)d_in[10];
  const float* bih10  = (const float*)d_in[11];
  const float* bhh10  = (const float*)d_in[12];
  const float* Wih11  = (const float*)d_in[13];
  const float* Whh11  = (const float*)d_in[14];
  const float* bih11  = (const float*)d_in[15];
  const float* bhh11  = (const float*)d_in[16];
  const float* ln_g   = (const float*)d_in[17];
  const float* ln_b   = (const float*)d_in[18];
  const float* W1     = (const float*)d_in[19];
  const float* b1     = (const float*)d_in[20];
  const float* W2     = (const float*)d_in[21];
  const float* b2     = (const float*)d_in[22];

  // Workspace: [ xpad bf16 B*T*32 | out0 bf16 B*T*2H | hT f32 2*B*H ]
  const size_t XPAD_BYTES = (size_t)Bn * Tn * XPAD * sizeof(bf16);     // 33.5 MB
  const size_t OUT0_BYTES = (size_t)Bn * Tn * (2 * Hn) * sizeof(bf16); // 134 MB
  bf16*  xpad = (bf16*)d_ws;
  bf16*  out0 = (bf16*)((char*)d_ws + XPAD_BYTES);
  float* hT   = (float*)((char*)d_ws + XPAD_BYTES + OUT0_BYTES);

  // 1) pad/convert x to bf16 [B,T,32]
  {
    const size_t total = (size_t)Bn * Tn * XPAD;
    pad_x_kernel<<<(unsigned)(total / 256), 256, 0, stream>>>(x, xpad);
  }

  dim3 gridRec(Bn / 16, 2);   // 16 batch tiles x 2 directions

  // 2) layer 0 recurrence (writes out0 sequence)
  gru_rec<1, true, false><<<gridRec, 128, 0, stream>>>(
      xpad, Wih00, Whh00, bih00, bhh00, Wih01, Whh01, bih01, bhh01,
      INn, out0, nullptr);

  // 3) layer 1 recurrence (writes final states only)
  gru_rec<4, false, true><<<gridRec, 128, 0, stream>>>(
      out0, Wih10, Whh10, bih10, bhh10, Wih11, Whh11, bih11, bhh11,
      2 * Hn, nullptr, hT);

  // 4) head
  gru_head<<<Bn, 128, 0, stream>>>(hT, ln_g, ln_b, W1, b1, W2, b2,
                                   (float*)d_out);
}
